// CKAN_10548439679188
// MI455X (gfx1250) — matrix operand order repaired and checked by reading, not compiled, not run on hardware
//
#include <hip/hip_runtime.h>
#include <hip/hip_bf16.h>

#define DIM   64
#define TT    32
#define BPB   4              // batch entries per block
#define ROWS  (BPB * TT)     // 128 rows per block
#define WAVES 8              // 256 threads / wave32

typedef __attribute__((ext_vector_type(16))) __bf16 v16bf;
typedef __attribute__((ext_vector_type(8)))  float  v8f;

// ---- dynamic LDS layout (bytes) ----
#define OFF_X    0        // bf16 X[128][128]   (reused later as f32 WOUT[128][64])
#define OFF_TE   32768    // f32  TE[128][64]
#define OFF_W1T  65536    // bf16 W1T[64][128]  (att_w1 transposed: [n][k])
#define OFF_G1T  81920    // bf16 G1T[64][128]
#define OFF_W2T  98304    // bf16 W2T[64][64]
#define OFF_G2T  106496   // bf16 G2T[64][64]
#define OFF_W3   114688   // f32  W3[64]
#define OFF_SS   114944   // bf16 SS[8 waves][16][64] per-wave scratch
#define OFF_SC   131328   // f32  SC[128] sigmoid scores
#define OFF_ATT  131840   // f32  ATT[128] softmax weights
#define SMEM_BYTES 132352

// A fragment (16x32 bf16, M x K) from row-major bf16 LDS [.., ldk]:
// lane m%16 = row; lane<16 -> K = k0..k0+15; lane>=16 -> K = k0+16..k0+31.
__device__ __forceinline__ v16bf load_A(const __bf16* base, int ldk, int row0,
                                        int k0, int lane) {
  const int m  = row0 + (lane & 15);
  const int kb = k0 + ((lane & 16) ? 16 : 0);
  const __bf16* p = base + m * ldk + kb;
  v16bf a;
#pragma unroll
  for (int i = 0; i < 8; ++i) a[i] = p[i];
#pragma unroll
  for (int i = 0; i < 8; ++i) a[8 + i] = p[8 + i];
  return a;
}

// B fragment (32x16 bf16, K x N) from N-major (transposed) weights WT[n][k]:
// lane n%16 = column; lane<16 -> K = k0..k0+15; lane>=16 -> K = k0+16..k0+31.
__device__ __forceinline__ v16bf load_B(const __bf16* WT, int ldk, int k0,
                                        int n0, int lane) {
  const int n  = n0 + (lane & 15);
  const int kb = k0 + ((lane & 16) ? 16 : 0);
  const __bf16* p = WT + n * ldk + kb;
  v16bf b;
#pragma unroll
  for (int i = 0; i < 16; ++i) b[i] = p[i];
  return b;
}

// One tower, one layer per blockIdx.y. Each block: 4 batch entries x 32 neighbors.
__global__ __launch_bounds__(256) void ckan_tower_kernel(
    const int* __restrict__ h_idx, const int* __restrict__ r_idx,
    const int* __restrict__ t_idx, const float* __restrict__ ent,
    const float* __restrict__ rel, const float* __restrict__ w1,
    const float* __restrict__ w2, const float* __restrict__ w3,
    const float* __restrict__ gw1, const float* __restrict__ gw2,
    float* __restrict__ out /* [L][B][DIM] */, int Bt) {
  extern __shared__ char smem[];
  __bf16* X   = (__bf16*)(smem + OFF_X);
  float*  TE  = (float* )(smem + OFF_TE);
  __bf16* W1T = (__bf16*)(smem + OFF_W1T);
  __bf16* G1T = (__bf16*)(smem + OFF_G1T);
  __bf16* W2T = (__bf16*)(smem + OFF_W2T);
  __bf16* G2T = (__bf16*)(smem + OFF_G2T);
  float*  W3  = (float* )(smem + OFF_W3);
  __bf16* SS  = (__bf16*)(smem + OFF_SS);
  float*  SC  = (float* )(smem + OFF_SC);
  float*  ATT = (float* )(smem + OFF_ATT);
  float*  WOUT = (float*)(smem + OFF_X);   // alias: X dead after A-frags cached

  const int tid = threadIdx.x;
  const int l   = blockIdx.y;
  const long long rbase = (long long)l * Bt * TT + (long long)blockIdx.x * ROWS;

  // ---- stage weights to LDS, transposed, bf16 ----
  for (int i = tid; i < 128 * 64; i += 256) {
    int k = i >> 6, d = i & 63;
    W1T[d * 128 + k] = (__bf16)w1[i];
    G1T[d * 128 + k] = (__bf16)gw1[i];
  }
  for (int i = tid; i < 64 * 64; i += 256) {
    int k = i >> 6, d = i & 63;
    W2T[d * 64 + k] = (__bf16)w2[i];
    G2T[d * 64 + k] = (__bf16)gw2[i];
  }
  if (tid < 64) W3[tid] = w3[tid];

  // ---- gather embeddings: X = [h|r] bf16, TE = t_e f32 (L2-resident) ----
  for (int i = tid; i < ROWS * DIM; i += 256) {
    int r = i >> 6, d = i & 63;
    long long hi = h_idx[rbase + r];
    long long ri = r_idx[rbase + r];
    long long ti = t_idx[rbase + r];
    X[r * 128 + d]      = (__bf16)ent[hi * DIM + d];
    X[r * 128 + 64 + d] = (__bf16)rel[ri * DIM + d];
    TE[r * DIM + d]     = ent[ti * DIM + d];
  }
  __syncthreads();

  const int wave = tid >> 5;
  const int lane = tid & 31;
  const int m0   = wave * 16;
  const int laneM = lane & 15;
  const int hiHalf = (lane >> 4) & 1;
  __bf16* S = SS + wave * 16 * 64;   // this wave's 16x64 scratch

  // ---- cache A fragments of X (reused by att + gate GEMMs) ----
  v16bf a[4];
#pragma unroll
  for (int k = 0; k < 4; ++k) a[k] = load_A(X, 128, m0, k * 32, lane);

  // ---- s1 = relu(X @ att_w1), K=128 N=64 : 16 WMMAs ----
  v8f acc[4];
#pragma unroll
  for (int n = 0; n < 4; ++n) {
    v8f c = {0.f, 0.f, 0.f, 0.f, 0.f, 0.f, 0.f, 0.f};
#pragma unroll
    for (int k = 0; k < 4; ++k) {
      v16bf b = load_B(W1T, 128, k * 32, n * 16, lane);
      c = __builtin_amdgcn_wmma_f32_16x16x32_bf16(false, a[k], false, b,
                                                  (short)0, c, false, false);
    }
    acc[n] = c;
  }
#pragma unroll
  for (int n = 0; n < 4; ++n)
#pragma unroll
    for (int r = 0; r < 8; ++r) {
      float v = acc[n][r];
      S[(r + (hiHalf ? 8 : 0)) * 64 + n * 16 + laneM] = (__bf16)(v > 0.f ? v : 0.f);
    }

  // ---- s2 = relu(s1 @ att_w2), K=64 : 8 WMMAs (wave-ordered LDS RAW) ----
  v16bf a2[2];
#pragma unroll
  for (int k = 0; k < 2; ++k) a2[k] = load_A(S, 64, 0, k * 32, lane);
#pragma unroll
  for (int n = 0; n < 4; ++n) {
    v8f c = {0.f, 0.f, 0.f, 0.f, 0.f, 0.f, 0.f, 0.f};
#pragma unroll
    for (int k = 0; k < 2; ++k) {
      v16bf b = load_B(W2T, 64, k * 32, n * 16, lane);
      c = __builtin_amdgcn_wmma_f32_16x16x32_bf16(false, a2[k], false, b,
                                                  (short)0, c, false, false);
    }
    acc[n] = c;
  }
#pragma unroll
  for (int n = 0; n < 4; ++n)
#pragma unroll
    for (int r = 0; r < 8; ++r) {
      float v = acc[n][r];
      S[(r + (hiHalf ? 8 : 0)) * 64 + n * 16 + laneM] = (__bf16)(v > 0.f ? v : 0.f);
    }

  // ---- s3 = sigmoid(s2 @ att_w3): per-row dot, lanes 0..15 own one row ----
  if (lane < 16) {
    float d3 = 0.f;
#pragma unroll 8
    for (int d = 0; d < DIM; ++d) d3 += (float)S[lane * 64 + d] * W3[d];
    SC[m0 + lane] = 1.f / (1.f + __expf(-d3));
  }

  // ---- g1 = relu(X @ gate_w1): 16 WMMAs (A frags cached) ----
#pragma unroll
  for (int n = 0; n < 4; ++n) {
    v8f c = {0.f, 0.f, 0.f, 0.f, 0.f, 0.f, 0.f, 0.f};
#pragma unroll
    for (int k = 0; k < 4; ++k) {
      v16bf b = load_B(G1T, 128, k * 32, n * 16, lane);
      c = __builtin_amdgcn_wmma_f32_16x16x32_bf16(false, a[k], false, b,
                                                  (short)0, c, false, false);
    }
    acc[n] = c;
  }
#pragma unroll
  for (int n = 0; n < 4; ++n)
#pragma unroll
    for (int r = 0; r < 8; ++r) {
      float v = acc[n][r];
      S[(r + (hiHalf ? 8 : 0)) * 64 + n * 16 + laneM] = (__bf16)(v > 0.f ? v : 0.f);
    }

  // ---- g2 = 2*sigmoid(g1 @ gate_w2): 8 WMMAs, keep in registers ----
  v8f g2[4];
#pragma unroll
  for (int k = 0; k < 2; ++k) a2[k] = load_A(S, 64, 0, k * 32, lane);
#pragma unroll
  for (int n = 0; n < 4; ++n) {
    v8f c = {0.f, 0.f, 0.f, 0.f, 0.f, 0.f, 0.f, 0.f};
#pragma unroll
    for (int k = 0; k < 2; ++k) {
      v16bf b = load_B(G2T, 64, k * 32, n * 16, lane);
      c = __builtin_amdgcn_wmma_f32_16x16x32_bf16(false, a2[k], false, b,
                                                  (short)0, c, false, false);
    }
    g2[n] = c;
  }

  __syncthreads();   // all SC written; X no longer read by anyone

  // ---- softmax over T=32 neighbors per batch entry ----
  if (tid < ROWS) {
    const int bl = tid >> 5;
    float mx = -1e30f;
#pragma unroll 8
    for (int t = 0; t < TT; ++t) mx = fmaxf(mx, SC[bl * TT + t]);
    float sum = 0.f;
#pragma unroll 8
    for (int t = 0; t < TT; ++t) sum += __expf(SC[bl * TT + t] - mx);
    ATT[tid] = __expf(SC[tid] - mx) / sum;
  }
  __syncthreads();

  // ---- weighted values att * gate * t_e into WOUT (reuses X space) ----
#pragma unroll
  for (int n = 0; n < 4; ++n)
#pragma unroll
    for (int r = 0; r < 8; ++r) {
      const int m = m0 + r + (hiHalf ? 8 : 0);
      const int d = n * 16 + laneM;
      const float g = 2.f / (1.f + __expf(-g2[n][r]));
      WOUT[m * DIM + d] = ATT[m] * g * TE[m * DIM + d];
    }
  __syncthreads();

  // ---- reduce over T=32 and store: one thread per (b_local, d) ----
  {
    const int bl = tid >> 6;      // 0..3
    const int d  = tid & 63;
    float s = 0.f;
#pragma unroll 8
    for (int t = 0; t < TT; ++t) s += WOUT[(bl * TT + t) * DIM + d];
    const long long b = (long long)blockIdx.x * BPB + bl;
    out[((long long)l * Bt + b) * DIM + d] = s;
  }
}

// Final scoring: origins + 192-dim dot + sigmoid. All gathers L2-resident.
__global__ __launch_bounds__(256) void ckan_score_kernel(
    const int* __restrict__ users, const int* __restrict__ items,
    const int* __restrict__ user_h, const float* __restrict__ ent,
    const float* __restrict__ ulay, const float* __restrict__ ilay,
    float* __restrict__ out, int Bt) {
  const int b = blockIdx.x * blockDim.x + threadIdx.x;
  if (b >= Bt) return;
  float uo[DIM];
#pragma unroll
  for (int d = 0; d < DIM; ++d) uo[d] = 0.f;
  for (int t = 0; t < TT; ++t) {
    const long long e = (long long)user_h[b * TT + t] * DIM;  // layer 0 slice
#pragma unroll
    for (int d = 0; d < DIM; ++d) uo[d] += ent[e + d];
  }
  const float* iv = ent + (long long)items[b] * DIM;
  float dot = 0.f;
#pragma unroll
  for (int d = 0; d < DIM; ++d) dot += (uo[d] * (1.f / TT)) * iv[d];
  for (int l = 0; l < 2; ++l) {
    const long long o = ((long long)l * Bt + b) * DIM;
#pragma unroll
    for (int d = 0; d < DIM; ++d) dot += ulay[o + d] * ilay[o + d];
  }
  out[b] = 1.f / (1.f + __expf(-dot));
}

extern "C" void kernel_launch(void* const* d_in, const int* in_sizes, int n_in,
                              void* d_out, int out_size, void* d_ws, size_t ws_size,
                              hipStream_t stream) {
  const int*   users  = (const int*)d_in[0];
  const int*   items  = (const int*)d_in[1];
  const int*   user_h = (const int*)d_in[2];
  const int*   user_r = (const int*)d_in[3];
  const int*   user_t = (const int*)d_in[4];
  const int*   item_h = (const int*)d_in[5];
  const int*   item_r = (const int*)d_in[6];
  const int*   item_t = (const int*)d_in[7];
  const float* ent    = (const float*)d_in[8];
  const float* rel    = (const float*)d_in[9];
  const float* w1     = (const float*)d_in[10];
  const float* w2     = (const float*)d_in[11];
  const float* w3     = (const float*)d_in[12];
  const float* gw1    = (const float*)d_in[13];
  const float* gw2    = (const float*)d_in[14];
  float* out = (float*)d_out;

  const int Bt = in_sizes[0];                 // 4096
  float* ulay = (float*)d_ws;                 // [2][B][64]
  float* ilay = ulay + (size_t)2 * Bt * DIM;  // [2][B][64]

  dim3 grid(Bt / BPB, 2);
  ckan_tower_kernel<<<grid, 256, SMEM_BYTES, stream>>>(
      user_h, user_r, user_t, ent, rel, w1, w2, w3, gw1, gw2, ulay, Bt);
  ckan_tower_kernel<<<grid, 256, SMEM_BYTES, stream>>>(
      item_h, item_r, item_t, ent, rel, w1, w2, w3, gw1, gw2, ilay, Bt);
  ckan_score_kernel<<<(Bt + 255) / 256, 256, 0, stream>>>(
      users, items, user_h, ent, ulay, ilay, out, Bt);
}
